// InteractionNet_49478023250672
// MI455X (gfx1250) — matrix operand order repaired and checked by reading, not compiled, not run on hardware
//
#include <hip/hip_runtime.h>
#include <cstdint>

// ---------------------------------------------------------------------------
// Types for gfx1250 WMMA (wave32): A/B = 16 bf16 per lane (8 VGPRs),
// C/D = 8 f32 per lane.
// ---------------------------------------------------------------------------
typedef __attribute__((ext_vector_type(16))) __bf16    v16bf;
typedef __attribute__((ext_vector_type(8)))  float     v8f;
typedef __attribute__((ext_vector_type(8)))  uint32_t  u8v;

#define N_TOTAL 100000
#define LDW 36   // u32 stride per q-row in padded LDS slice (128B data + 16B pad)

__device__ __forceinline__ uint32_t f32_to_bf16_bits(float f) {
  union { float f; uint32_t u; } v; v.f = f;
  uint32_t u = v.u;
  return ((u + 0x7FFFu + ((u >> 16) & 1u)) >> 16) & 0xFFFFu;
}
__device__ __forceinline__ uint32_t pack2(float a, float b) {
  return f32_to_bf16_bits(a) | (f32_to_bf16_bits(b) << 16);
}

__device__ __forceinline__ v8f wmma_bf16(v16bf a, v16bf b, v8f c) {
  // D = A(16x32) x B(32x16) + C, f32 accumulate
  return __builtin_amdgcn_wmma_f32_16x16x32_bf16(
      /*neg_a=*/false, a, /*neg_b=*/false, b,
      /*c_mod=*/(short)0, c, /*reuse_a=*/false, /*reuse_b=*/false);
}

// CDNA5 async copy: global -> LDS without touching VGPR data, tracked by
// ASYNCcnt.  GVS addressing: uniform SGPR64 base + per-lane 32-bit byte offset.
// VDST VGPR carries the per-lane LDS byte offset (low 32 bits of the flat
// LDS-aperture address == wave-relative LDS offset).
__device__ __forceinline__ void async_b128(uint32_t lds_off, const uint32_t* base,
                                           uint32_t byte_off) {
  asm volatile("global_load_async_to_lds_b128 %0, %1, %2"
               :: "v"(lds_off), "v"(byte_off), "s"(base)
               : "memory");
}
__device__ __forceinline__ void wait_async() {
  asm volatile("s_wait_asynccnt 0" ::: "memory");
}
__device__ __forceinline__ uint32_t lds_addr(const void* p) {
  return (uint32_t)(uintptr_t)p;
}
// padded LDS byte offset for uint4-chunk index i (0..511) of an 8KB slice
__device__ __forceinline__ uint32_t pad_off(int i) {
  return (uint32_t)((i >> 3) * (LDW * 4) + (i & 7) * 16);
}

// A-fragment (16x32 bf16, ISA layout): lane<16 holds row M=lane, K = kbase+{0..7,16..23};
// lane>=16 holds same row set, K = kbase+{8..15,24..31}.
__device__ __forceinline__ v16bf a_frag(const float* __restrict__ rowp, int kbase, int hi) {
  const int o = kbase + hi * 8;
  u8v u;
#pragma unroll
  for (int j = 0; j < 4; ++j) {
    u[j]     = pack2(rowp[o + 2 * j],      rowp[o + 2 * j + 1]);
    u[4 + j] = pack2(rowp[o + 16 + 2 * j], rowp[o + 16 + 2 * j + 1]);
  }
  return __builtin_bit_cast(v16bf, u);
}

// B-fragment (32x16 bf16) from padded LDS slice laid out [q][k] (k fast, bf16).
// lane<16: column N=lane, K = 32c+{0..7,16..23}; lane>=16: column N=lane-16, K = 32c+{8..15,24..31}.
__device__ __forceinline__ v16bf b_frag(const uint32_t* slice, int t, int c, int ln, int hi) {
  const uint32_t* p = slice + (t * 16 + ln) * LDW + c * 16 + hi * 4;
  uint4 a = *(const uint4*)(p);
  uint4 b = *(const uint4*)(p + 8);
  u8v u = { a.x, a.y, a.z, a.w, b.x, b.y, b.z, b.w };
  return __builtin_bit_cast(v16bf, u);
}

// ---------------------------------------------------------------------------
// Prep: transpose + bf16-convert weights once.  Outputs (u32 = 2 packed bf16):
//   W0T   [p][q][k]   = lin0_w[k][64+64p+q]     (64*64*64 bf16 = 512KB)
//   W0c0T [q][k]      = lin0_w[k][q]            (8KB)
//   B0T   [q][p]      = lin0_b[64+64p+q]        (8KB)
//   W1T   [q][k]      = lin1_w[k][1+q]          (8KB)
//   w1c0  [k]         = lin1_w[k][0]  (f32)     (256B)
// ---------------------------------------------------------------------------
__global__ void prep_kernel(const float* __restrict__ w0, const float* __restrict__ b0,
                            const float* __restrict__ w1,
                            uint32_t* __restrict__ W0T, uint32_t* __restrict__ W0c0T,
                            uint32_t* __restrict__ B0T, uint32_t* __restrict__ W1T,
                            float* __restrict__ w1c0) {
  const int i = blockIdx.x * blockDim.x + threadIdx.x;
  if (i < 64 * 64 * 32) {  // W0T pairs: p, q, k2
    const int k2 = i & 31, q = (i >> 5) & 63, p = i >> 11;
    const float* c = w0 + 64 + p * 64 + q;           // column, row stride 4160
    W0T[i] = pack2(c[(size_t)(2 * k2) * 4160], c[(size_t)(2 * k2 + 1) * 4160]);
  }
  if (i < 64 * 32) {       // W0c0T pairs: q, k2
    const int k2 = i & 31, q = i >> 5;
    const float* c = w0 + q;
    W0c0T[i] = pack2(c[(size_t)(2 * k2) * 4160], c[(size_t)(2 * k2 + 1) * 4160]);
  }
  if (i < 64 * 32) {       // B0T pairs: q, p2
    const int p2 = i & 31, q = i >> 5;
    B0T[i] = pack2(b0[64 + (2 * p2) * 64 + q], b0[64 + (2 * p2 + 1) * 64 + q]);
  }
  if (i < 64 * 32) {       // W1T pairs: q, k2
    const int k2 = i & 31, q = i >> 5;
    W1T[i] = pack2(w1[(2 * k2) * 65 + 1 + q], w1[(2 * k2 + 1) * 65 + 1 + q]);
  }
  if (i < 64) w1c0[i] = w1[i * 65];
}

// ---------------------------------------------------------------------------
// Fused hypernetwork: 8 waves/block, 16 samples/wave, 128 samples/block.
// acc[n,q] = (int_x@W0[:, :64])[n,q] + b0[q]
//          + sum_p x[n,p] * (int_x@W0[:,64+64p:+64])[n,q]   (p loop, WMMA)
//          + (x @ B0mat)[n,q]                               (WMMA, folds b0 weight part)
// h = relu(acc);  out[n] = b1[0] + int_x[n]·w1c0 + sum_q h[n,q]*(b1[1+q] + (int_x@W1)[n,q])
// ---------------------------------------------------------------------------
__global__ __launch_bounds__(256) void hyper_fused(
    const float* __restrict__ int_x, const float* __restrict__ x,
    const float* __restrict__ lin0_b, const float* __restrict__ lin1_b,
    const uint32_t* __restrict__ W0T, const uint32_t* __restrict__ W0c0T,
    const uint32_t* __restrict__ B0T, const uint32_t* __restrict__ W1T,
    const float* __restrict__ w1c0, float* __restrict__ out) {
  __shared__ uint32_t sw0[2][64 * LDW];   // W0 slice double buffer
  __shared__ uint32_t sc[3][64 * LDW];    // W0c0T, B0T, W1T
  __shared__ float    sx[8][16 * 65];     // x tile, row-major with pad
  __shared__ float    sb0[64];
  __shared__ float    sb1[66];
  __shared__ float    sw1c0[64];

  const int tid = threadIdx.x;
  const int wid = tid >> 5, lane = tid & 31;
  const int ln = lane & 15, hi = lane >> 4;

  // ---- async-stage slice 0 + the three constant 64x64 matrices ----
  const int i0 = tid * 2, i1 = tid * 2 + 1;
  const uint32_t l0 = pad_off(i0), l1 = pad_off(i1);
  const uint32_t g0 = (uint32_t)i0 * 16u, g1 = (uint32_t)i1 * 16u;
  const uint32_t lds_w0[2] = { lds_addr(&sw0[0][0]), lds_addr(&sw0[1][0]) };
  async_b128(lds_w0[0] + l0, W0T, g0);
  async_b128(lds_w0[0] + l1, W0T, g1);
  async_b128(lds_addr(&sc[0][0]) + l0, W0c0T, g0);
  async_b128(lds_addr(&sc[0][0]) + l1, W0c0T, g1);
  async_b128(lds_addr(&sc[1][0]) + l0, B0T, g0);
  async_b128(lds_addr(&sc[1][0]) + l1, B0T, g1);
  async_b128(lds_addr(&sc[2][0]) + l0, W1T, g0);
  async_b128(lds_addr(&sc[2][0]) + l1, W1T, g1);

  if (tid < 64) { sb0[tid] = lin0_b[tid]; sw1c0[tid] = w1c0[tid]; }
  if (tid < 65) sb1[tid] = lin1_b[tid];

  size_t rowbase = (size_t)blockIdx.x * 128 + (size_t)wid * 16;
  if (rowbase > (size_t)(N_TOTAL - 16)) rowbase = N_TOTAL - 16;  // N % 16 == 0

  for (int i = lane; i < 16 * 64; i += 32) {      // stage x tile for scalar reads
    const int n = i >> 6, k = i & 63;
    sx[wid][n * 65 + k] = x[(rowbase + n) * 64 + k];
  }

  const float* ir = int_x + (rowbase + ln) * 64;
  const float* xr = x     + (rowbase + ln) * 64;
  const v16bf ai0 = a_frag(ir, 0, hi), ai1 = a_frag(ir, 32, hi);
  const v16bf ax0 = a_frag(xr, 0, hi), ax1 = a_frag(xr, 32, hi);

  wait_async();
  __syncthreads();

  const v8f vzero = {0.f, 0.f, 0.f, 0.f, 0.f, 0.f, 0.f, 0.f};
  v8f acc[4];
#pragma unroll
  for (int t = 0; t < 4; ++t)    // bias-generating GEMM initializes acc
    acc[t] = wmma_bf16(ai1, b_frag(sc[0], t, 1, ln, hi),
             wmma_bf16(ai0, b_frag(sc[0], t, 0, ln, hi), vzero));

  for (int p = 0; p < 64; ++p) {
    // async double-buffer refill: copy slice p+1 straight into LDS while we
    // compute on slice p; no VGPR staging, tracked by ASYNCcnt.
    if (p + 1 < 64) {
      const uint32_t nxt = lds_w0[(p + 1) & 1];
      const uint32_t gofs = (uint32_t)(p + 1) * 8192u;
      async_b128(nxt + l0, W0T, gofs + g0);
      async_b128(nxt + l1, W0T, gofs + g1);
    }
    if (p + 2 < 64) __builtin_prefetch(W0T + (size_t)(p + 2) * 2048, 0, 1);

    const uint32_t* cur = sw0[p & 1];
    float xs[8];
#pragma unroll
    for (int vi = 0; vi < 8; ++vi) xs[vi] = sx[wid][(vi + 8 * hi) * 65 + p];

#pragma unroll
    for (int t = 0; t < 4; ++t) {
      v8f T = wmma_bf16(ai1, b_frag(cur, t, 1, ln, hi),
              wmma_bf16(ai0, b_frag(cur, t, 0, ln, hi), vzero));
#pragma unroll
      for (int vi = 0; vi < 8; ++vi) acc[t][vi] += xs[vi] * T[vi];
    }
    wait_async();
    __syncthreads();
  }

  // epilogue: + b0[q], + x@B0mat (WMMA accumulates into C), relu
#pragma unroll
  for (int t = 0; t < 4; ++t) {
    const float bq = sb0[t * 16 + ln];
#pragma unroll
    for (int vi = 0; vi < 8; ++vi) acc[t][vi] += bq;
    acc[t] = wmma_bf16(ax1, b_frag(sc[1], t, 1, ln, hi),
             wmma_bf16(ax0, b_frag(sc[1], t, 0, ln, hi), acc[t]));
#pragma unroll
    for (int vi = 0; vi < 8; ++vi) acc[t][vi] = fmaxf(acc[t][vi], 0.0f);
  }

  // layer 1: w2 = int_x@W1 (WMMA), out partial = sum_q h*(w2 + b1[1+q])
  float part[8];
#pragma unroll
  for (int vi = 0; vi < 8; ++vi) part[vi] = 0.0f;
#pragma unroll
  for (int t = 0; t < 4; ++t) {
    v8f W = wmma_bf16(ai1, b_frag(sc[2], t, 1, ln, hi),
            wmma_bf16(ai0, b_frag(sc[2], t, 0, ln, hi), vzero));
    const float b1q = sb1[1 + t * 16 + ln];
#pragma unroll
    for (int vi = 0; vi < 8; ++vi) part[vi] += acc[t][vi] * (W[vi] + b1q);
  }
#pragma unroll
  for (int m = 1; m <= 8; m <<= 1) {
#pragma unroll
    for (int vi = 0; vi < 8; ++vi) part[vi] += __shfl_xor(part[vi], m, 32);
  }

  if (ln < 8) {
    const size_t row = rowbase + ln + 8 * hi;
    const float* irr = int_x + row * 64;
    float d = sb1[0];
#pragma unroll
    for (int k = 0; k < 64; ++k) d += irr[k] * sw1c0[k];
    if (row < N_TOTAL) out[row] = part[ln] + d;
  }
}

extern "C" void kernel_launch(void* const* d_in, const int* in_sizes, int n_in,
                              void* d_out, int out_size, void* d_ws, size_t ws_size,
                              hipStream_t stream) {
  (void)in_sizes; (void)n_in; (void)out_size; (void)ws_size;
  const float* int_x = (const float*)d_in[0];
  const float* x     = (const float*)d_in[1];
  const float* w0    = (const float*)d_in[2];
  const float* b0    = (const float*)d_in[3];
  const float* w1    = (const float*)d_in[4];
  const float* b1    = (const float*)d_in[5];
  float* out = (float*)d_out;

  uint8_t* ws = (uint8_t*)d_ws;
  uint32_t* W0T   = (uint32_t*)(ws);              // 524288 B
  uint32_t* W0c0T = (uint32_t*)(ws + 524288);     // 8192 B
  uint32_t* B0T   = (uint32_t*)(ws + 532480);     // 8192 B
  uint32_t* W1T   = (uint32_t*)(ws + 540672);     // 8192 B
  float*    w1c0  = (float*)(ws + 548864);        // 256 B

  prep_kernel<<<512, 256, 0, stream>>>(w0, b0, w1, W0T, W0c0T, B0T, W1T, w1c0);

  const int nblocks = (N_TOTAL + 127) / 128;      // 782
  hyper_fused<<<nblocks, 256, 0, stream>>>(int_x, x, b0, b1,
                                           W0T, W0c0T, B0T, W1T, w1c0, out);
}
